// TSAdaptivePatcher_53042846106007
// MI455X (gfx1250) — matrix operand order repaired and checked by reading, not compiled, not run on hardware
//
#include <hip/hip_runtime.h>

typedef __attribute__((ext_vector_type(16))) _Float16 v16h;
typedef __attribute__((ext_vector_type(8)))  float    v8f;

#define B_   64
#define C_   64
#define S_   8192
#define NP_  512
#define P_   16
#define PT_  8               // patches per tile
#define ROW_ (PT_*16 + 16)   // padded LDS row (floats) per channel: 144 dwords

// ---------------------------------------------------------------------------
// Block transpose: out[b, p, c, k] = x[b, c, p*16+k]
// One block per (b, tile of 8 patches). LDS-staged so that global loads are
// contiguous per channel (8*64B runs) and global stores are contiguous per
// patch (64*64B runs). LDS row padded by 16 dwords -> conflict-free b128 DS.
// ---------------------------------------------------------------------------
__global__ __launch_bounds__(256)
void patch_transpose_kernel(const float* __restrict__ x, float* __restrict__ out) {
    __shared__ __align__(16) float lds[C_ * ROW_];   // 36 KiB

    const int t  = threadIdx.x;
    const int b  = blockIdx.y;
    const int p0 = blockIdx.x * PT_;

    // ---- load phase: for each channel c, the 8-patch row is 512B contiguous
    const float4* __restrict__ in4 = (const float4*)x;
    const int r  = t & 31;        // float4 index within the c-row (32 x 16B)
    const int cl = t >> 5;        // 0..7
    #pragma unroll
    for (int c = cl; c < C_; c += 8) {
        long gi = (long)(b * C_ + c) * (S_ / 4) + (long)p0 * 4 + r;
        float4 v = in4[gi];
        *(float4*)&lds[c * ROW_ + r * 4] = v;
    }
    __syncthreads();

    // ---- store phase: for each patch p, the 64-channel row is 4KB contiguous
    float4* __restrict__ out4 = (float4*)out;
    const int k4 = t & 3;         // float4 within a 16-float k run
    const int c  = t >> 2;        // 0..63
    #pragma unroll
    for (int p = 0; p < PT_; ++p) {
        float4 v = *(const float4*)&lds[c * ROW_ + p * 16 + k4 * 4];
        long go = ((long)(b * NP_ + p0 + p) * C_ + c) * 4 + k4;
        out4[go] = v;
    }
}

// ---------------------------------------------------------------------------
// Padding mask via WMMA: each wave computes 32 window sums with one
// v_wmma_f32_16x16x32_f16.
//   A[m, k]  = mask[b, tt*512 + m*32 + k]        (16 rows x 32 values)
//   B[k, n]  = (k < 16 && n == 0) || (k >= 16 && n == 1)
//   D[m, 0]  = sum of window 2m, D[m, 1] = sum of window 2m+1
// A k order used by this code (contiguous, 16x32): lanes 0-15 -> M=lane, K = h      (0..15)
//                                 lanes16-31 -> M=lane-16, K = h+16   (16..31)
// B k order used by this code (same as A):          lanes 0-15 -> N=lane,    K = h      (0..15)
//                                 lanes16-31 -> N=lane-16, K = h+16   (16..31)
// ---------------------------------------------------------------------------
__global__ __launch_bounds__(32)
void padding_mask_kernel(const int* __restrict__ mask, float* __restrict__ outm) {
    const int tile = blockIdx.x;      // 64 b * 16 tiles
    const int b    = tile >> 4;
    const int tt   = tile & 15;       // tile covers windows [tt*32, tt*32+32)
    const int lane = threadIdx.x;     // 0..31
    const int m    = lane & 15;       // row M of A / col N of B
    const bool hi  = lane >= 16;

    const int* __restrict__ arow = mask + (long)b * S_ + tt * 512 + m * 32;

    // B: only (N=0, K<16) and (N=1, K>=16) are 1.0 -> lanes 0 and 17 all-ones
    const _Float16 bval = ((!hi && m == 0) || (hi && m == 1))
                              ? (_Float16)1.0f : (_Float16)0.0f;
    v16h a, bb;
    #pragma unroll
    for (int h = 0; h < 16; ++h) {
        const int K = hi ? h + 16 : h;
        a[h]  = (_Float16)(float)arow[K];   // 0.0 or 1.0, exact in f16
        bb[h] = bval;
    }

    v8f c = {};
    // (neg_a, A, neg_b, B, c_mod, C, reuse_a, reuse_b)
    c = __builtin_amdgcn_wmma_f32_16x16x32_f16(false, a, false, bb,
                                               (short)0, c, false, false);

    // D layout: VGPR r, lanes 0-15 -> (M=r, N=lane); lanes 16-31 -> (M=r+8)
    if (m < 2) {
        #pragma unroll
        for (int r = 0; r < 8; ++r) {
            const int M = r + (hi ? 8 : 0);
            const int w = tt * 32 + 2 * M + m;      // window index in [0,512)
            outm[(long)b * NP_ + w] = (c[r] >= 8.0f) ? 1.0f : 0.0f;
        }
    }
}

// ---------------------------------------------------------------------------
extern "C" void kernel_launch(void* const* d_in, const int* in_sizes, int n_in,
                              void* d_out, int out_size, void* d_ws, size_t ws_size,
                              hipStream_t stream) {
    const float* x    = (const float*)d_in[0];   // [64, 64, 8192] f32
    const int*   mask = (const int*)d_in[1];     // [64, 8192] i32
    float* out  = (float*)d_out;                              // patches: 64*512*1024
    float* outm = out + (long)B_ * NP_ * (C_ * P_);           // padding mask: 64*512

    dim3 tgrid(NP_ / PT_, B_);                   // 64 x 64 blocks
    patch_transpose_kernel<<<tgrid, 256, 0, stream>>>(x, out);
    padding_mask_kernel<<<B_ * 16, 32, 0, stream>>>(mask, outm);
}
